// Transformer_20392504722105
// MI455X (gfx1250) — matrix operand order repaired and checked by reading, not compiled, not run on hardware
//
#include <hip/hip_runtime.h>

#define B_    32
#define NE_   512
#define HID_  1024
#define HEADS_ 16
#define DH_   64

typedef __attribute__((ext_vector_type(16))) _Float16 v16h;
typedef __attribute__((ext_vector_type(8)))  _Float16 v8h;
typedef __attribute__((ext_vector_type(8)))  float    v8f;

union AFrag { v16h v; v8h h[2]; };

__device__ __forceinline__ v8f wmma_f16(v16h a, v16h b, v8f c) {
  // D = A(16x32 f16) * B(32x16 f16) + C(16x16 f32)
  return __builtin_amdgcn_wmma_f32_16x16x32_f16(false, a, false, b, (short)0, c,
                                                false, false);
}

// ---------------------------------------------------------------------------
// Weight transpose + f32->f16 convert:  W[K][N] -> WT[N][K]
// ---------------------------------------------------------------------------
__global__ __launch_bounds__(256)
void wtrans(const float* __restrict__ W, _Float16* __restrict__ WT, int K, int N) {
  __shared__ float t[32][33];
  const int n0 = blockIdx.x * 32, k0 = blockIdx.y * 32;
  for (int i = threadIdx.y; i < 32; i += 8)
    t[i][threadIdx.x] = W[(size_t)(k0 + i) * N + n0 + threadIdx.x];
  __syncthreads();
  for (int i = threadIdx.y; i < 32; i += 8)
    WT[(size_t)(n0 + i) * K + k0 + threadIdx.x] = (_Float16)t[threadIdx.x][i];
}

// ---------------------------------------------------------------------------
// LayerNorm 1: x (f32, row=1024) -> f16
// ---------------------------------------------------------------------------
__global__ __launch_bounds__(256)
void ln_to_f16(const float* __restrict__ x, const float* __restrict__ g,
               const float* __restrict__ bta, _Float16* __restrict__ out) {
  const int row = blockIdx.x, tid = threadIdx.x;
  const float* xr = x + (size_t)row * HID_;
  float v[4], s = 0.f, s2 = 0.f;
  #pragma unroll
  for (int i = 0; i < 4; ++i) { v[i] = xr[tid + 256 * i]; s += v[i]; s2 += v[i] * v[i]; }
  __shared__ float rs[256], rq[256];
  rs[tid] = s; rq[tid] = s2; __syncthreads();
  for (int off = 128; off; off >>= 1) {
    if (tid < off) { rs[tid] += rs[tid + off]; rq[tid] += rq[tid + off]; }
    __syncthreads();
  }
  const float mean = rs[0] * (1.f / HID_);
  const float var  = rq[0] * (1.f / HID_) - mean * mean;
  const float rinv = rsqrtf(var + 1e-12f);
  #pragma unroll
  for (int i = 0; i < 4; ++i) {
    const int c = tid + 256 * i;
    out[(size_t)row * HID_ + c] = (_Float16)((v[i] - mean) * rinv * g[c] + bta[c]);
  }
}

// ---------------------------------------------------------------------------
// WMMA GEMM: A[M=16384][1024] f16  x  WT[N][1024] f16  (+bias)
// MODE 0: N=2048, de-interleave into Q,K  (b,h,ne,dh) f16
// MODE 1: N=1024, store V transposed      (b,h,dh,ne) f16
// MODE 2: N=1024, xo = xres + A*W + bias  (f32)
// Block = 128 threads (4 waves), block tile 64(M) x 64(N), wave tile 16x64.
// ---------------------------------------------------------------------------
template <int MODE>
__global__ __launch_bounds__(128)
void gemm_wmma(const _Float16* __restrict__ A, const _Float16* __restrict__ WT,
               const float* __restrict__ bias,
               _Float16* __restrict__ o0, _Float16* __restrict__ o1,
               const float* __restrict__ xres, float* __restrict__ xo) {
  const int lane = threadIdx.x & 31;
  const int wave = threadIdx.x >> 5;
  const int n0 = blockIdx.x * 64;
  const int m0 = blockIdx.y * 64 + wave * 16;
  const int kn = lane & 15, kg = lane >> 4;

  const _Float16* ap = A + (size_t)(m0 + kn) * HID_ + kg * 16;
  const _Float16* bp[4];
  int col[4];
  #pragma unroll
  for (int nt = 0; nt < 4; ++nt) {
    col[nt] = n0 + nt * 16 + kn;
    bp[nt] = WT + (size_t)col[nt] * HID_ + kg * 16;
  }
  v8f acc[4];
  #pragma unroll
  for (int nt = 0; nt < 4; ++nt) acc[nt] = v8f{0, 0, 0, 0, 0, 0, 0, 0};

  for (int k0 = 0; k0 < HID_; k0 += 32) {
    AFrag a;
    a.h[0] = *(const v8h*)(ap + k0);
    a.h[1] = *(const v8h*)(ap + k0 + 8);
    #pragma unroll
    for (int nt = 0; nt < 4; ++nt)
      acc[nt] = wmma_f16(a.v, *(const v16h*)(bp[nt] + k0), acc[nt]);
  }

  const int mrow = m0 + 8 * kg;           // first of 8 rows this lane owns
  const int b = mrow >> 9;                // /NE_
  const int ne = mrow & (NE_ - 1);

  #pragma unroll
  for (int nt = 0; nt < 4; ++nt) {
    const int c = col[nt];
    const float bb = bias[c];
    if (MODE == 0) {
      const int h = c >> 7, d = (c >> 1) & 63, isk = c & 1;
      _Float16* dst = (isk ? o1 : o0) +
                      ((size_t)(b * HEADS_ + h) * NE_ + ne) * DH_ + d;
      #pragma unroll
      for (int r = 0; r < 8; ++r) dst[(size_t)r * DH_] = (_Float16)(acc[nt][r] + bb);
    } else if (MODE == 1) {
      const int h = c >> 6, d = c & 63;
      v8h hv;
      #pragma unroll
      for (int r = 0; r < 8; ++r) hv[r] = (_Float16)(acc[nt][r] + bb);
      *(v8h*)(o0 + ((size_t)(b * HEADS_ + h) * DH_ + d) * NE_ + ne) = hv;
    } else {
      #pragma unroll
      for (int r = 0; r < 8; ++r) {
        const size_t idx = (size_t)(mrow + r) * HID_ + c;
        xo[idx] = xres[idx] + acc[nt][r] + bb;
      }
    }
  }
}

// ---------------------------------------------------------------------------
// Attention: one wave per (b, h, 16 q-rows).
// S = (Q K^T)/8 - (1-m)*1e10 in LDS -> masked softmax -> P f16 -> P*V (WMMA)
// ---------------------------------------------------------------------------
#define SPS 516   // f32 S row stride (pad: avoid bank conflicts)
#define SPP 520   // f16 P row stride (pad, keeps 16B alignment)

__global__ __launch_bounds__(32)
void attn(const _Float16* __restrict__ Q, const _Float16* __restrict__ Kb,
          const _Float16* __restrict__ Vt, const float* __restrict__ mask,
          _Float16* __restrict__ att) {
  __shared__ float S[16 * SPS];
  __shared__ __align__(16) _Float16 P[16 * SPP];
  __shared__ float Msh[NE_];

  const int lane = threadIdx.x;
  const int wg = blockIdx.x;                 // b*512 + h*32 + qt
  const int qt = wg & 31, h = (wg >> 5) & 15, b = wg >> 9;
  const int bh = b * HEADS_ + h;
  const int kn = lane & 15, kg = lane >> 4;

  for (int i = lane; i < NE_; i += 32) Msh[i] = mask[(size_t)b * NE_ + i];
  __syncthreads();

  // Q fragments for this 16-row tile (K-dim = DH = 64 -> 2 chunks of 32)
  const _Float16* qp = Q + ((size_t)bh * NE_ + qt * 16 + kn) * DH_ + kg * 16;
  AFrag aq0, aq1;
  aq0.h[0] = *(const v8h*)(qp);       aq0.h[1] = *(const v8h*)(qp + 8);
  aq1.h[0] = *(const v8h*)(qp + 32);  aq1.h[1] = *(const v8h*)(qp + 40);

  // Pass 1: scores into LDS
  for (int kt = 0; kt < 32; ++kt) {
    const _Float16* kp = Kb + ((size_t)bh * NE_ + kt * 16 + kn) * DH_ + kg * 16;
    v8f s = v8f{0, 0, 0, 0, 0, 0, 0, 0};
    s = wmma_f16(aq0.v, *(const v16h*)(kp), s);
    s = wmma_f16(aq1.v, *(const v16h*)(kp + 32), s);
    const float pen = (1.0f - Msh[kt * 16 + kn]) * 1e10f;
    #pragma unroll
    for (int r = 0; r < 8; ++r)
      S[(8 * kg + r) * SPS + kt * 16 + kn] = s[r] * 0.125f - pen;
  }
  __syncthreads();

  // Pass 2: masked softmax (2 lanes per row, 256 keys each)
  {
    const int row = lane >> 1, hs = lane & 1;
    const float* srow = &S[row * SPS + hs * 256];
    const float* mrow = &Msh[hs * 256];
    _Float16* prow = &P[row * SPP + hs * 256];
    float mx = -3.0e38f;
    for (int i = 0; i < 256; ++i) mx = fmaxf(mx, srow[i]);
    mx = fmaxf(mx, __shfl_xor(mx, 1, 32));
    float sum = 0.f;
    for (int i = 0; i < 256; ++i) {
      const float p = mrow[i] * __expf(srow[i] - mx);
      sum += p;
      prow[i] = (_Float16)p;
    }
    sum += __shfl_xor(sum, 1, 32);
    const float inv = 1.0f / (sum + 1e-10f);
    for (int i = 0; i < 256; ++i) prow[i] = (_Float16)((float)prow[i] * inv);
  }
  __syncthreads();

  // Pass 3: att = P * V   (A from LDS, B from transposed V, contiguous)
  v8f acc[4];
  #pragma unroll
  for (int nt = 0; nt < 4; ++nt) acc[nt] = v8f{0, 0, 0, 0, 0, 0, 0, 0};
  for (int kc = 0; kc < NE_; kc += 32) {
    AFrag a;
    const _Float16* pp = &P[kn * SPP + kc + kg * 16];
    a.h[0] = *(const v8h*)(pp);
    a.h[1] = *(const v8h*)(pp + 8);
    #pragma unroll
    for (int nt = 0; nt < 4; ++nt) {
      const _Float16* vp =
          Vt + ((size_t)bh * DH_ + nt * 16 + kn) * NE_ + kc + kg * 16;
      acc[nt] = wmma_f16(a.v, *(const v16h*)(vp), acc[nt]);
    }
  }
  #pragma unroll
  for (int nt = 0; nt < 4; ++nt) {
    const int d = h * 64 + nt * 16 + kn;
    const int ne0 = qt * 16 + 8 * kg;
    #pragma unroll
    for (int r = 0; r < 8; ++r)
      att[((size_t)b * NE_ + ne0 + r) * HID_ + d] = (_Float16)acc[nt][r];
  }
}

// ---------------------------------------------------------------------------
// LayerNorm 2 (in place, f32) then multiply row by its mask value.
// ---------------------------------------------------------------------------
__global__ __launch_bounds__(256)
void ln2_mask(float* __restrict__ xo, const float* __restrict__ g,
              const float* __restrict__ bta, const float* __restrict__ mask) {
  const int row = blockIdx.x, tid = threadIdx.x;
  float* xr = xo + (size_t)row * HID_;
  float v[4], s = 0.f, s2 = 0.f;
  #pragma unroll
  for (int i = 0; i < 4; ++i) { v[i] = xr[tid + 256 * i]; s += v[i]; s2 += v[i] * v[i]; }
  __shared__ float rs[256], rq[256];
  rs[tid] = s; rq[tid] = s2; __syncthreads();
  for (int off = 128; off; off >>= 1) {
    if (tid < off) { rs[tid] += rs[tid + off]; rq[tid] += rq[tid + off]; }
    __syncthreads();
  }
  const float mean = rs[0] * (1.f / HID_);
  const float var  = rq[0] * (1.f / HID_) - mean * mean;
  const float rinv = rsqrtf(var + 1e-12f);
  const float mk = mask[row];
  #pragma unroll
  for (int i = 0; i < 4; ++i) {
    const int c = tid + 256 * i;
    xr[c] = ((v[i] - mean) * rinv * g[c] + bta[c]) * mk;
  }
}

// ---------------------------------------------------------------------------
// Masked mean-pool: out[b][c] = sum_ne xo / (sum mask + 1e-5)
// ---------------------------------------------------------------------------
__global__ __launch_bounds__(256)
void pool(const float* __restrict__ xo, const float* __restrict__ mask,
          float* __restrict__ out) {
  const int b = blockIdx.x;
  const int c = blockIdx.y * 256 + threadIdx.x;
  __shared__ float cr[256];
  cr[threadIdx.x] = mask[(size_t)b * NE_ + threadIdx.x] +
                    mask[(size_t)b * NE_ + 256 + threadIdx.x];
  __syncthreads();
  for (int off = 128; off; off >>= 1) {
    if (threadIdx.x < off) cr[threadIdx.x] += cr[threadIdx.x + off];
    __syncthreads();
  }
  const float cnt = cr[0];
  float s = 0.f;
  for (int ne = 0; ne < NE_; ++ne) s += xo[((size_t)b * NE_ + ne) * HID_ + c];
  out[(size_t)b * HID_ + c] = s / (cnt + 1e-5f);
}

// ---------------------------------------------------------------------------
extern "C" void kernel_launch(void* const* d_in, const int* in_sizes, int n_in,
                              void* d_out, int out_size, void* d_ws, size_t ws_size,
                              hipStream_t stream) {
  (void)in_sizes; (void)n_in; (void)out_size; (void)ws_size;
  const float* x     = (const float*)d_in[0];
  const float* mask  = (const float*)d_in[1];
  const float* Wqk   = (const float*)d_in[2];
  const float* bqk   = (const float*)d_in[3];
  const float* Wv    = (const float*)d_in[4];
  const float* bv    = (const float*)d_in[5];
  const float* Wpost = (const float*)d_in[6];
  const float* bpost = (const float*)d_in[7];
  const float* g1    = (const float*)d_in[8];
  const float* b1    = (const float*)d_in[9];
  const float* g2    = (const float*)d_in[10];
  const float* b2    = (const float*)d_in[11];

  char* ws = (char*)d_ws;
  const size_t MiB = 1024 * 1024;
  _Float16* WqkT = (_Float16*)(ws + 0);          //  4 MiB  (2048x1024 f16)
  _Float16* WvT  = (_Float16*)(ws + 4 * MiB);    //  2 MiB
  _Float16* WpT  = (_Float16*)(ws + 6 * MiB);    //  2 MiB
  _Float16* inp  = (_Float16*)(ws + 8 * MiB);    // 32 MiB (reused as att)
  _Float16* Qb   = (_Float16*)(ws + 40 * MiB);   // 32 MiB
  _Float16* Kbuf = (_Float16*)(ws + 72 * MiB);   // 32 MiB
  _Float16* Vtb  = (_Float16*)(ws + 104 * MiB);  // 32 MiB
  float*    xo   = (float*)(ws + 40 * MiB);      // 64 MiB, reuses Q+K after attn

  wtrans<<<dim3(2 * HID_ / 32, HID_ / 32), dim3(32, 8), 0, stream>>>(Wqk, WqkT, HID_, 2 * HID_);
  wtrans<<<dim3(HID_ / 32, HID_ / 32), dim3(32, 8), 0, stream>>>(Wv, WvT, HID_, HID_);
  wtrans<<<dim3(HID_ / 32, HID_ / 32), dim3(32, 8), 0, stream>>>(Wpost, WpT, HID_, HID_);

  ln_to_f16<<<B_ * NE_, 256, 0, stream>>>(x, g1, b1, inp);

  gemm_wmma<0><<<dim3(2 * HID_ / 64, B_ * NE_ / 64), 128, 0, stream>>>(
      inp, WqkT, bqk, Qb, Kbuf, nullptr, nullptr);
  gemm_wmma<1><<<dim3(HID_ / 64, B_ * NE_ / 64), 128, 0, stream>>>(
      inp, WvT, bv, Vtb, nullptr, nullptr, nullptr);

  attn<<<B_ * HEADS_ * (NE_ / 16), 32, 0, stream>>>(Qb, Kbuf, Vtb, mask, inp);

  gemm_wmma<2><<<dim3(HID_ / 64, B_ * NE_ / 64), 128, 0, stream>>>(
      inp, WpT, bpost, nullptr, nullptr, x, xo);

  ln2_mask<<<B_ * NE_, 256, 0, stream>>>(xo, g2, b2, mask);
  pool<<<dim3(B_, HID_ / 256), 256, 0, stream>>>(xo, mask, (float*)d_out);
}
